// CrossAttentionT2S_7722351198257
// MI455X (gfx1250) — matrix operand order repaired and checked by reading, not compiled, not run on hardware
//
#include <hip/hip_runtime.h>
#include <hip/hip_bf16.h>
#include <string.h>

// ---------------------------------------------------------------------------
// CDNA5 (gfx1250) wave32 WMMA implementation of CrossAttentionT2S.
//   B=8, H=12, DH=64, DIM=768, AP=VP=196, TS=T=8 -> L = 196*8 = 1568 tokens,
//   M = B*L = 12544 rows for all projections. SCALE = 1/8.
//   Data movement: GLOBAL_LOAD_ASYNC_TO_LDS_B128 + double-buffered LDS tiles,
//   partial s_wait_asynccnt waits for load/compute overlap.
// ---------------------------------------------------------------------------

typedef __attribute__((ext_vector_type(16))) __bf16 v16bf;
typedef __attribute__((ext_vector_type(8)))  float  v8f;

#define LTOK   1568          // tokens per batch element (196*8)
#define MROWS  12544         // 8 * 1568
#define DIMK   768

// ---- CDNA5 async global->LDS path (guarded; sync fallback keeps compile OK)
#if defined(__has_builtin)
#  if __has_builtin(__builtin_amdgcn_global_load_async_to_lds_b128)
#    define HAVE_ASYNC_LDS 1
#  else
#    define HAVE_ASYNC_LDS 0
#  endif
#else
#  define HAVE_ASYNC_LDS 0
#endif

#if HAVE_ASYNC_LDS
// Builtin signature (from hipcc diagnostic): 16-byte int-vector pointers,
// global (AS1) source and LDS (AS3) destination, plus imm offset / cpol.
typedef int v4i_t __attribute__((vector_size(16)));
typedef __attribute__((address_space(1))) v4i_t as1_v4i;
typedef __attribute__((address_space(3))) v4i_t as3_v4i;
__device__ __forceinline__ void g2l_b128(const void* g, void* l) {
    __builtin_amdgcn_global_load_async_to_lds_b128((as1_v4i*)g, (as3_v4i*)l, 0, 0);
}
#  if __has_builtin(__builtin_amdgcn_s_wait_asynccnt)
#    define WAIT_ASYNC(n) __builtin_amdgcn_s_wait_asynccnt(n)
#  else
#    define WAIT_ASYNC(n) asm volatile("s_wait_asynccnt " #n ::: "memory")
#  endif
#else
__device__ __forceinline__ void g2l_b128(const void* g, void* l) {
    *(uint4*)l = *(const uint4*)g;
}
#  define WAIT_ASYNC(n) ((void)0)
#endif

#if defined(__has_builtin) && __has_builtin(__builtin_amdgcn_s_wait_dscnt)
#  define WAIT_DS0() __builtin_amdgcn_s_wait_dscnt(0)
#else
#  define WAIT_DS0() asm volatile("s_wait_dscnt 0x0" ::: "memory")
#endif

union FragU {
    v16bf v;
    unsigned int   u[8];
    unsigned short s16[16];
};

__device__ __forceinline__ unsigned short f2bf(float f) {
    union { float f; unsigned int u; } x{f};
    unsigned int r = x.u + 0x7FFFu + ((x.u >> 16) & 1u);   // round-nearest-even
    return (unsigned short)(r >> 16);
}

// A-fragment K index for 16-bit 16x32 A tiles (ISA 7.12.2), pair index v (0..7),
// lane group g = lane>>4.  Returns even K of the bf16 pair {K, K+1}.
__device__ __forceinline__ int afrag_k(int v, int g) {
    return (v < 4) ? (g * 8 + v * 2) : (16 + g * 8 + (v - 4) * 2);
}

// ---------------------------------------------------------------------------
// 1) Fused positional embedding + fp32 -> bf16.
// ---------------------------------------------------------------------------
__global__ void embed_kernel(const float* __restrict__ src,
                             const float* __restrict__ spos,
                             const float* __restrict__ tpos,
                             unsigned short* __restrict__ dst,
                             long srcOff) {
    size_t idx = (size_t)blockIdx.x * blockDim.x + threadIdx.x;
    if (idx >= (size_t)MROWS * DIMK) return;
    int d      = (int)(idx % DIMK);
    size_t row = idx / DIMK;               // b*1568 + n*8 + t
    int b  = (int)(row / LTOK);
    int rr = (int)(row % LTOK);
    int n  = rr >> 3;
    int t  = rr & 7;
    float v = src[(size_t)srcOff + (size_t)n * (64 * DIMK) + (size_t)(b * 8 + t) * DIMK + d]
            + spos[(size_t)n * DIMK + d] + tpos[(size_t)t * DIMK + d];
    dst[idx] = f2bf(v);
}

// ---------------------------------------------------------------------------
// 2) Weight transpose to K x N bf16:  Wt[k*N + n] = W[n*K + k],  K = 768.
// ---------------------------------------------------------------------------
__global__ void transw_kernel(const float* __restrict__ W,
                              unsigned short* __restrict__ Wt, int N) {
    size_t idx = (size_t)blockIdx.x * blockDim.x + threadIdx.x;
    if (idx >= (size_t)N * DIMK) return;
    int n = (int)(idx / DIMK);
    int k = (int)(idx % DIMK);
    Wt[(size_t)k * N + n] = f2bf(W[idx]);
}

// ---------------------------------------------------------------------------
// 3) WMMA GEMM:  C(M x N) = alpha * (A(M x 768) * Bt(768 x N) + bias)
//    128 threads (4 waves), block tile 64x64, K-step 32, double-buffered LDS
//    filled by async global->LDS copies (4 per wave per K-step -> wait(4)).
//    mode 0: bf16 out (row-major M x N)    mode 1: f32 scatter into d_out.
// ---------------------------------------------------------------------------
__global__ __launch_bounds__(128)
void gemm_kernel(const unsigned short* __restrict__ A,
                 const unsigned short* __restrict__ Bt,
                 const float* __restrict__ bias,
                 float alpha,
                 unsigned short* __restrict__ Cb,
                 float* __restrict__ Cf,
                 int N, int mode) {
    __shared__ unsigned short Ash[2][64 * 32];
    __shared__ unsigned short Bsh[2][32 * 64];
    const int tid  = threadIdx.x;
    const int wave = tid >> 5;
    const int lane = tid & 31;
    const int g    = lane >> 4;
    const int l16  = lane & 15;
    const int m0   = blockIdx.y * 64;
    const int n0   = blockIdx.x * 64;

    auto stage = [&](int buf, int k0) {
        int r  = tid >> 2, cb  = (tid & 3) * 8;       // A: 64x32
        g2l_b128(A + (size_t)(m0 + r)      * DIMK + k0 + cb, Ash[buf] + r * 32 + cb);
        g2l_b128(A + (size_t)(m0 + r + 32) * DIMK + k0 + cb, Ash[buf] + (r + 32) * 32 + cb);
        int r2 = tid >> 3, cb2 = (tid & 7) * 8;       // B: 32x64
        g2l_b128(Bt + (size_t)(k0 + r2)      * N + n0 + cb2, Bsh[buf] + r2 * 64 + cb2);
        g2l_b128(Bt + (size_t)(k0 + r2 + 16) * N + n0 + cb2, Bsh[buf] + (r2 + 16) * 64 + cb2);
    };

    v8f acc[4] = {};
    stage(0, 0);

    for (int ks = 0; ks < DIMK / 32; ++ks) {
        const int cur = ks & 1;
        if (ks + 1 < DIMK / 32) {
            stage(cur ^ 1, (ks + 1) * 32);            // prefetch next tile async
            if (ks + 2 < DIMK / 32)                   // L2 prefetch one slab ahead
                __builtin_prefetch(A + (size_t)(m0 + (tid >> 2)) * DIMK + (ks + 2) * 32, 0, 1);
            WAIT_ASYNC(4);                            // only current buffer must be done
        } else {
            WAIT_ASYNC(0);
        }
        __syncthreads();

        // ---- A fragment: 16x32, lane=M within wave tile ----
        FragU af;
        const int arow = wave * 16 + l16;
        #pragma unroll
        for (int v = 0; v < 8; ++v)
            af.u[v] = *(const unsigned int*)(Ash[cur] + arow * 32 + afrag_k(v, g));

        // ---- 4 B fragments (lane = N column, 16 K per lane in table order) ----
        #pragma unroll
        for (int nc = 0; nc < 4; ++nc) {
            FragU bu;
            #pragma unroll
            for (int e = 0; e < 16; ++e)
                bu.s16[e] = Bsh[cur][(afrag_k(e >> 1, g) + (e & 1)) * 64 + nc * 16 + l16];
            v16bf bv = bu.v;
            acc[nc] = __builtin_amdgcn_wmma_f32_16x16x32_bf16(
                false, af.v, false, bv, (short)0, acc[nc], false, false);
        }
        __syncthreads();   // close WAR hazard before buffer is re-staged
    }

    // ---- epilogue ----
    #pragma unroll
    for (int nc = 0; nc < 4; ++nc) {
        int n  = n0 + nc * 16 + l16;
        float bvv = bias[n];
        #pragma unroll
        for (int r = 0; r < 8; ++r) {
            int m = m0 + wave * 16 + r + 8 * g;
            float val = alpha * (acc[nc][r] + bvv);
            if (mode == 0) {
                Cb[(size_t)m * N + n] = f2bf(val);
            } else {
                // m = b*1568 + na*8 + t  ->  out row = na*64 + b*8 + t
                int b = m / LTOK, rr = m % LTOK, na = rr >> 3, t = rr & 7;
                Cf[((size_t)na * 64 + b * 8 + t) * DIMK + n] = val;
            }
        }
    }
}

// ---------------------------------------------------------------------------
// 4) Flash-attention per (b, h, 16-query tile).  One wave32 per block.
//    Double-buffered async K/V staging (16 async ops per tile -> wait(16)).
// ---------------------------------------------------------------------------
__global__ __launch_bounds__(32)
void attn_kernel(const unsigned short* __restrict__ Qm,
                 const unsigned short* __restrict__ KVm,
                 unsigned short* __restrict__ Om) {
    __shared__ unsigned short Ksh[2][32 * 64];
    __shared__ unsigned short Vsh[2][32 * 64];
    __shared__ unsigned short Psh[16 * 32];
    const int lane = threadIdx.x;
    const int qt   = blockIdx.x;          // 0..97
    const int h    = blockIdx.y;          // 0..11
    const int b    = blockIdx.z;          // 0..7
    const int g    = lane >> 4;
    const int l16  = lane & 15;
    const size_t rowBase = (size_t)b * LTOK;

    auto stageKV = [&](int buf, int kbase) {
        const unsigned short* kr = KVm + (rowBase + kbase + lane) * 1536 + h * 64;
        #pragma unroll
        for (int j = 0; j < 8; ++j) {
            g2l_b128(kr + j * 8,       Ksh[buf] + lane * 64 + j * 8);
            g2l_b128(kr + 768 + j * 8, Vsh[buf] + lane * 64 + j * 8);
        }
    };

    // ---- Q A-fragments (two 16x32 chunks over head-dim 64) ----
    FragU qa[2];
    {
        const unsigned short* qrow = Qm + (rowBase + qt * 16 + l16) * DIMK + h * 64;
        #pragma unroll
        for (int c = 0; c < 2; ++c)
            #pragma unroll
            for (int v = 0; v < 8; ++v)
                qa[c].u[v] = *(const unsigned int*)(qrow + c * 32 + afrag_k(v, g));
    }

    float mrow[8], lrow[8];
    #pragma unroll
    for (int r = 0; r < 8; ++r) { mrow[r] = -3.0e38f; lrow[r] = 0.f; }
    v8f oacc[4] = {};

    stageKV(0, 0);

    for (int it = 0; it < 49; ++it) {
        const int cur = it & 1;
        if (it + 1 < 49) {
            WAIT_DS0();                       // prior ds_loads of other buffer done
            stageKV(cur ^ 1, (it + 1) * 32);  // async prefetch of next K/V tile
            WAIT_ASYNC(16);                   // current tile's 16 copies complete
        } else {
            WAIT_ASYNC(0);
        }
        __syncthreads();   // single-wave workgroup: S_NOP + compiler fence

        // ---- S = Q K^T : 2 key chunks x 2 contraction chunks ----
        v8f s[2] = {};
        #pragma unroll
        for (int nc = 0; nc < 2; ++nc) {
            #pragma unroll
            for (int c = 0; c < 2; ++c) {
                FragU bf_;  // B frag: lane = key within chunk, elems = 16 d (table order)
                #pragma unroll
                for (int e = 0; e < 16; ++e)
                    bf_.s16[e] = Ksh[cur][(nc * 16 + l16) * 64 + c * 32 + afrag_k(e >> 1, g) + (e & 1)];
                s[nc] = __builtin_amdgcn_wmma_f32_16x16x32_bf16(
                    false, qa[c].v, false, bf_.v, (short)0, s[nc], false, false);
            }
        }

        // ---- online softmax (rows r+8g live on this lane's acc slots) ----
        #pragma unroll
        for (int r = 0; r < 8; ++r) {
            float mx = fmaxf(s[0][r], s[1][r]);
            #pragma unroll
            for (int off = 8; off >= 1; off >>= 1)
                mx = fmaxf(mx, __shfl_xor(mx, off, 32));
            float mnew = fmaxf(mrow[r], mx);
            float corr = __expf(mrow[r] - mnew);
            float p0 = __expf(s[0][r] - mnew);
            float p1 = __expf(s[1][r] - mnew);
            float ps = p0 + p1;
            #pragma unroll
            for (int off = 8; off >= 1; off >>= 1)
                ps += __shfl_xor(ps, off, 32);
            lrow[r] = lrow[r] * corr + ps;
            mrow[r] = mnew;
            #pragma unroll
            for (int nc = 0; nc < 4; ++nc) oacc[nc][r] *= corr;
            int m = r + 8 * g;
            Psh[m * 32 + l16]      = f2bf(p0);
            Psh[m * 32 + 16 + l16] = f2bf(p1);
        }
        __syncthreads();

        // ---- P A-fragment from LDS bounce ----
        FragU pa;
        #pragma unroll
        for (int v = 0; v < 8; ++v)
            pa.u[v] = *(const unsigned int*)(Psh + l16 * 32 + afrag_k(v, g));

        // ---- O += P * V ----
        #pragma unroll
        for (int nc = 0; nc < 4; ++nc) {
            FragU vu;
            #pragma unroll
            for (int e = 0; e < 16; ++e)
                vu.s16[e] = Vsh[cur][(afrag_k(e >> 1, g) + (e & 1)) * 64 + nc * 16 + l16];
            v16bf bv = vu.v;
            oacc[nc] = __builtin_amdgcn_wmma_f32_16x16x32_bf16(
                false, pa.v, false, bv, (short)0, oacc[nc], false, false);
        }
        __syncthreads();
    }

    // ---- normalize and store bf16 ----
    #pragma unroll
    for (int nc = 0; nc < 4; ++nc) {
        #pragma unroll
        for (int r = 0; r < 8; ++r) {
            int m = qt * 16 + r + 8 * g;
            float val = oacc[nc][r] / lrow[r];
            Om[(rowBase + m) * DIMK + h * 64 + nc * 16 + l16] = f2bf(val);
        }
    }
}

// ---------------------------------------------------------------------------
// Host launcher
// ---------------------------------------------------------------------------
extern "C" void kernel_launch(void* const* d_in, const int* in_sizes, int n_in,
                              void* d_out, int out_size, void* d_ws, size_t ws_size,
                              hipStream_t stream) {
    (void)in_sizes; (void)n_in; (void)out_size; (void)ws_size;
    const float* s_x    = (const float*)d_in[0];
    const float* t_x    = (const float*)d_in[1];
    const float* c_spos = (const float*)d_in[2];
    const float* v_spos = (const float*)d_in[3];
    const float* c_tpos = (const float*)d_in[4];
    const float* v_tpos = (const float*)d_in[5];
    const float* q_w    = (const float*)d_in[6];
    const float* q_b    = (const float*)d_in[7];
    const float* kv_w   = (const float*)d_in[8];
    const float* kv_b   = (const float*)d_in[9];
    const float* proj_w = (const float*)d_in[10];
    const float* proj_b = (const float*)d_in[11];
    float* out = (float*)d_out;

    char* ws = (char*)d_ws;
    const size_t SZ_MD = (size_t)MROWS * DIMK * sizeof(unsigned short);   // 19.27 MB
    unsigned short* Sm   = (unsigned short*)(ws);                          // queries-in
    unsigned short* TPm  = (unsigned short*)(ws + SZ_MD);                  // kv-in
    unsigned short* Qm   = (unsigned short*)(ws + 2 * SZ_MD);
    unsigned short* Om   = (unsigned short*)(ws + 3 * SZ_MD);
    unsigned short* KVm  = (unsigned short*)(ws + 4 * SZ_MD);              // M x 1536
    char* wbase = ws + 4 * SZ_MD + (size_t)MROWS * 1536 * sizeof(unsigned short);
    unsigned short* WtQ  = (unsigned short*)(wbase);
    unsigned short* WtKV = (unsigned short*)(wbase + (size_t)DIMK * DIMK * 2);
    unsigned short* WtP  = (unsigned short*)(wbase + (size_t)DIMK * DIMK * 2
                                                   + (size_t)DIMK * 1536 * 2);

    const float SCALE = 0.125f;   // 64^-0.5

    // 1) embeddings (t_x[1:] -> offset of one leading row = 64*768 floats)
    {
        size_t total = (size_t)MROWS * DIMK;
        int grid = (int)((total + 255) / 256);
        embed_kernel<<<grid, 256, 0, stream>>>(s_x, c_spos, c_tpos, Sm, 0L);
        embed_kernel<<<grid, 256, 0, stream>>>(t_x, v_spos, v_tpos, TPm, (long)(64 * DIMK));
    }
    // 2) weight transposes
    {
        int grid768  = (int)(((size_t)DIMK * DIMK + 255) / 256);
        int grid1536 = (int)(((size_t)1536 * DIMK + 255) / 256);
        transw_kernel<<<grid768,  256, 0, stream>>>(q_w,    WtQ,  DIMK);
        transw_kernel<<<grid1536, 256, 0, stream>>>(kv_w,   WtKV, 1536);
        transw_kernel<<<grid768,  256, 0, stream>>>(proj_w, WtP,  DIMK);
    }
    // 3) Q projection (scale folded) and KV projection
    gemm_kernel<<<dim3(DIMK / 64, MROWS / 64), 128, 0, stream>>>(
        Sm, WtQ, q_b, SCALE, Qm, nullptr, DIMK, 0);
    gemm_kernel<<<dim3(1536 / 64, MROWS / 64), 128, 0, stream>>>(
        TPm, WtKV, kv_b, 1.0f, KVm, nullptr, 1536, 0);
    // 4) attention
    attn_kernel<<<dim3(LTOK / 16, 12, 8), 32, 0, stream>>>(Qm, KVm, Om);
    // 5) output projection + scatter 'b (n t) d -> n (b t) d'
    gemm_kernel<<<dim3(DIMK / 64, MROWS / 64), 128, 0, stream>>>(
        Om, WtP, proj_b, 1.0f, nullptr, out, DIMK, 1);
}
